// Layer_90245852824215
// MI455X (gfx1250) — matrix operand order repaired and checked by reading, not compiled, not run on hardware
//
#include <hip/hip_runtime.h>
#include <hip/hip_bf16.h>

typedef __bf16 bf16_t;
typedef __bf16 v16bf __attribute__((ext_vector_type(16)));
typedef float  v8f   __attribute__((ext_vector_type(8)));

union AFrag {
    v16bf v;
    uint4 u[2];   // 32 bytes
};

// ---------------------------------------------------------------------------
// CDNA5 async global->LDS copy (16B per lane), tracked by ASYNCcnt.
// Flat pointer to __shared__ has the workgroup-relative LDS byte offset in its
// low 32 bits (ISA 10.2 aperture mapping), which is what VDST wants.
// ---------------------------------------------------------------------------
__device__ __forceinline__ void async_copy16(const bf16_t* g, bf16_t* l) {
    unsigned loff = (unsigned)(unsigned long long)l;
    asm volatile("global_load_async_to_lds_b128 %0, %1, off"
                 :: "v"(loff), "v"(g) : "memory");
}
__device__ __forceinline__ void async_wait0() {
    asm volatile("s_wait_asynccnt 0" ::: "memory");
}

// ---------------------------------------------------------------------------
// fp32 -> bf16 transposed cast: in[R][C] -> out[C][R]
// (weights pre-transposed so GEMM B tiles are contiguous along K -> pure
//  async b128 staging and transpose-free B-fragment reads)
// ---------------------------------------------------------------------------
__global__ void cast_transpose_f32_bf16(const float* __restrict__ in,
                                        bf16_t* __restrict__ out, int R, int C) {
    int idx = blockIdx.x * blockDim.x + threadIdx.x;
    if (idx < R * C) {
        int r = idx / C, c = idx % C;
        out[(size_t)c * R + r] = (bf16_t)in[idx];
    }
}

// ---------------------------------------------------------------------------
// LayerNorm (ddof=1, eps=1e-6), fp32 in -> bf16 out. One row (768) per block.
// ---------------------------------------------------------------------------
__global__ __launch_bounds__(256) void layernorm_kernel(
    const float* __restrict__ x, const float* __restrict__ gamma,
    const float* __restrict__ beta, bf16_t* __restrict__ out)
{
    const int D = 768;
    const int row = blockIdx.x;
    const float* xr = x + (size_t)row * D;

    float v[3];
    float s = 0.f, ss = 0.f;
#pragma unroll
    for (int i = 0; i < 3; ++i) {
        v[i] = xr[threadIdx.x + i * 256];
        s += v[i];
        ss += v[i] * v[i];
    }
    __shared__ float rs[256], rss[256];
    rs[threadIdx.x] = s; rss[threadIdx.x] = ss;
    __syncthreads();
    for (int off = 128; off > 0; off >>= 1) {
        if (threadIdx.x < off) {
            rs[threadIdx.x]  += rs[threadIdx.x + off];
            rss[threadIdx.x] += rss[threadIdx.x + off];
        }
        __syncthreads();
    }
    const float mean = rs[0] / (float)D;
    const float var  = (rss[0] - mean * rs[0]) / (float)(D - 1);  // unbiased
    const float rstd = rsqrtf(var + 1e-6f);

    bf16_t* orow = out + (size_t)row * D;
#pragma unroll
    for (int i = 0; i < 3; ++i) {
        int c = threadIdx.x + i * 256;
        orow[c] = (bf16_t)(gamma[c] * ((v[i] - mean) * rstd) + beta[c]);
    }
}

// ---------------------------------------------------------------------------
// bf16 WMMA GEMM: C[M,N] = ascale * (A[M,K] * Bt[N,K]^T)  (+bias)(relu)(+res)
// BM=128, BN=128, BK=32; 256 threads = 8 waves (2 M x 4 N), 64x32 per wave.
// Double-buffered LDS fed by async global->LDS b128 copies.
// ---------------------------------------------------------------------------
constexpr int BM = 128, BN = 128, BK = 32;
constexpr int LDA_S = BK + 8;   // padded LDS stride (halfwords)
constexpr int LDB_S = BK + 8;

template<bool HAS_BIAS, bool RELU, bool HAS_RES, bool OUT_BF16>
__global__ __launch_bounds__(256) void gemm_bf16_kernel(
    const bf16_t* __restrict__ A, const bf16_t* __restrict__ Bt,
    const float* __restrict__ bias, const float* __restrict__ res,
    void* __restrict__ Cout, int M, int N, int K, float ascale)
{
    __shared__ __align__(16) bf16_t As[2 * BM * LDA_S];
    __shared__ __align__(16) bf16_t Bs[2 * BN * LDB_S];

    const int tid   = threadIdx.x;
    const int lane  = tid & 31;
    const int wave  = tid >> 5;
    const int waveM = wave >> 2;     // 0..1
    const int waveN = wave & 3;      // 0..3
    const int half  = lane >> 4;     // 0/1
    const int l16   = lane & 15;

    const int m0 = blockIdx.x * BM;
    const int n0 = blockIdx.y * BN;

    v8f acc[4][2] = {};

    auto stage = [&](int k0, int buf) {
        bf16_t* Asb = As + buf * (BM * LDA_S);
        bf16_t* Bsb = Bs + buf * (BN * LDB_S);
#pragma unroll
        for (int i = 0; i < 2; ++i) {
            int c = tid + i * 256;          // 0..511 chunks of 8 halfwords
            int row  = c >> 2;
            int col8 = (c & 3) * 8;
            async_copy16(A  + (size_t)(m0 + row) * K + k0 + col8,
                         Asb + row * LDA_S + col8);
            async_copy16(Bt + (size_t)(n0 + row) * K + k0 + col8,
                         Bsb + row * LDB_S + col8);
        }
    };

    stage(0, 0);
    async_wait0();
    __syncthreads();

    int buf = 0;
    for (int k0 = 0; k0 < K; k0 += BK) {
        if (k0 + BK < K) stage(k0 + BK, buf ^ 1);

        const bf16_t* Asb = As + buf * (BM * LDA_S);
        const bf16_t* Bsb = Bs + buf * (BN * LDB_S);

        AFrag af[4];
#pragma unroll
        for (int mi = 0; mi < 4; ++mi) {
            const bf16_t* p = Asb + (waveM * 64 + mi * 16 + l16) * LDA_S;
            af[mi].u[0] = *(const uint4*)(p + 8 * half);        // K = 8h .. 8h+7
            af[mi].u[1] = *(const uint4*)(p + 16 + 8 * half);   // K = 16+8h .. 23+8h
        }
        AFrag bfr[2];
#pragma unroll
        for (int ni = 0; ni < 2; ++ni) {
            const bf16_t* p = Bsb + (waveN * 32 + ni * 16 + l16) * LDB_S;
            bfr[ni].u[0] = *(const uint4*)(p + 8 * half);       // K = 8h .. 8h+7
            bfr[ni].u[1] = *(const uint4*)(p + 16 + 8 * half);
        }
#pragma unroll
        for (int mi = 0; mi < 4; ++mi)
#pragma unroll
            for (int ni = 0; ni < 2; ++ni)
                acc[mi][ni] = __builtin_amdgcn_wmma_f32_16x16x32_bf16(
                    false, af[mi].v, false, bfr[ni].v, (short)0, acc[mi][ni], false, false);

        async_wait0();
        __syncthreads();
        buf ^= 1;
    }

    // epilogue
#pragma unroll
    for (int mi = 0; mi < 4; ++mi) {
#pragma unroll
        for (int ni = 0; ni < 2; ++ni) {
            int col = n0 + waveN * 32 + ni * 16 + l16;
            float bv = HAS_BIAS ? bias[col] : 0.f;
#pragma unroll
            for (int r = 0; r < 8; ++r) {
                int row = m0 + waveM * 64 + mi * 16 + r + 8 * half;
                float v = acc[mi][ni][r] * ascale + bv;
                if (RELU) v = v > 0.f ? v : 0.f;
                if (HAS_RES) v += res[(size_t)row * N + col];
                if (OUT_BF16) ((bf16_t*)Cout)[(size_t)row * N + col] = (bf16_t)v;
                else          ((float*)Cout)[(size_t)row * N + col] = v;
            }
        }
    }
}

// ---------------------------------------------------------------------------
// Flash attention: one wave per (b, h, 16-query tile). hs=64, key chunks of 32.
// Q comes in pre-scaled by 1/sqrt(hs) (folded into the Q-GEMM epilogue).
// Causal mask applied only on the diagonal chunk. Per-wave LDS staging;
// K chunk staged with async copies (per-wave ASYNCcnt wait, no barrier).
// ---------------------------------------------------------------------------
constexpr int ATT_WPB      = 4;
constexpr int KLDS_STRIDE  = 72;   // 32 rows  (K chunk, row-major, hs contiguous)
constexpr int VLDS_STRIDE  = 40;   // 64 rows  (V transposed: [hs][key])
constexpr int PLDS_STRIDE  = 40;   // 16 rows  (P: [q][key])
constexpr int WAVE_LDS = 32 * KLDS_STRIDE + 64 * VLDS_STRIDE + 16 * PLDS_STRIDE;

__global__ __launch_bounds__(128) void attention_kernel(
    const bf16_t* __restrict__ Q, const bf16_t* __restrict__ Km,
    const bf16_t* __restrict__ V, bf16_t* __restrict__ Out)
{
    __shared__ __align__(16) bf16_t smem[ATT_WPB * WAVE_LDS];

    const int lane = threadIdx.x & 31;
    const int wv   = threadIdx.x >> 5;
    const int gw   = blockIdx.x * ATT_WPB + wv;
    const int qt   = gw & 127;            // S/16 = 128 tiles
    const int h    = (gw >> 7) % 12;
    const int b    = gw / (128 * 12);
    const int q0   = qt * 16;
    const int half = lane >> 4, l16 = lane & 15;

    bf16_t* Klds = smem + wv * WAVE_LDS;
    bf16_t* Vt   = Klds + 32 * KLDS_STRIDE;
    bf16_t* Pl   = Vt   + 64 * VLDS_STRIDE;

    const bf16_t* qbase = Q  + ((size_t)b * 2048) * 768 + h * 64;
    const bf16_t* kbase = Km + ((size_t)b * 2048) * 768 + h * 64;
    const bf16_t* vbase = V  + ((size_t)b * 2048) * 768 + h * 64;

    // resident Q A-fragments (hs chunks 0-31, 32-63)
    AFrag qf[2];
#pragma unroll
    for (int f = 0; f < 2; ++f) {
        const bf16_t* p = qbase + (size_t)(q0 + l16) * 768 + f * 32;
        qf[f].u[0] = *(const uint4*)(p + 8 * half);
        qf[f].u[1] = *(const uint4*)(p + 16 + 8 * half);
    }

    v8f acc[4] = {};
    float m_i[8], l_i[8];
#pragma unroll
    for (int r = 0; r < 8; ++r) { m_i[r] = -3.0e38f; l_i[r] = 0.f; }

    const int nchunks = (q0 + 16 + 31) >> 5;
    for (int kc = 0; kc < nchunks; ++kc) {
        const int kb0 = kc * 32;
        if (kc + 1 < nchunks)   // speculative prefetch of next K chunk
            __builtin_prefetch(kbase + (size_t)(kb0 + 32) * 768, 0, 0);

        // stage K chunk [32][64] via async global->LDS copies
#pragma unroll
        for (int i = 0; i < 8; ++i) {
            int c = lane + i * 32;
            int row = c >> 3, part = (c & 7) * 8;
            async_copy16(kbase + (size_t)(kb0 + row) * 768 + part,
                         Klds + row * KLDS_STRIDE + part);
        }
        // stage V transposed: Vt[hs][key]
#pragma unroll
        for (int i = 0; i < 64; ++i) {
            int e = i * 32 + lane;
            int k = e >> 6, j = e & 63;
            Vt[j * VLDS_STRIDE + k] = vbase[(size_t)(kb0 + k) * 768 + j];
        }
        async_wait0();   // per-wave: our K copies have landed

        // scores: S[16x32] via 2 key-blocks x 2 hs-chunks
        v8f s[2] = {};
#pragma unroll
        for (int kb = 0; kb < 2; ++kb) {
#pragma unroll
            for (int c = 0; c < 2; ++c) {
                AFrag bfrg;
                const bf16_t* p = Klds + (kb * 16 + l16) * KLDS_STRIDE + c * 32;
                bfrg.u[0] = *(const uint4*)(p + 8 * half);
                bfrg.u[1] = *(const uint4*)(p + 16 + 8 * half);
                s[kb] = __builtin_amdgcn_wmma_f32_16x16x32_bf16(
                    false, qf[c].v, false, bfrg.v, (short)0, s[kb], false, false);
            }
        }
        // causal mask: only the diagonal chunk can straddle the boundary
        if (kb0 + 31 > q0) {
#pragma unroll
            for (int kb = 0; kb < 2; ++kb) {
                int key = kb0 + kb * 16 + l16;
#pragma unroll
                for (int r = 0; r < 8; ++r) {
                    int qrow = q0 + r + 8 * half;
                    if (key > qrow) s[kb][r] = -3.0e38f;
                }
            }
        }
        // online softmax (row = r + 8*half lives across a 16-lane group)
#pragma unroll
        for (int r = 0; r < 8; ++r) {
            float cand = fmaxf(s[0][r], s[1][r]);
#pragma unroll
            for (int off = 8; off > 0; off >>= 1)
                cand = fmaxf(cand, __shfl_xor(cand, off, 32));
            float mnew  = fmaxf(m_i[r], cand);
            float alpha = __expf(m_i[r] - mnew);
            m_i[r] = mnew;
            float p0 = __expf(s[0][r] - mnew);
            float p1 = __expf(s[1][r] - mnew);
            float ls = p0 + p1;
#pragma unroll
            for (int off = 8; off > 0; off >>= 1)
                ls += __shfl_xor(ls, off, 32);
            l_i[r] = l_i[r] * alpha + ls;
#pragma unroll
            for (int c = 0; c < 4; ++c) acc[c][r] *= alpha;
            int m = r + 8 * half;
            Pl[m * PLDS_STRIDE + l16]      = (bf16_t)p0;
            Pl[m * PLDS_STRIDE + 16 + l16] = (bf16_t)p1;
        }
        // P C-layout -> A-layout via LDS round-trip
        AFrag pf;
        {
            const bf16_t* p = Pl + l16 * PLDS_STRIDE;
            pf.u[0] = *(const uint4*)(p + 8 * half);
            pf.u[1] = *(const uint4*)(p + 16 + 8 * half);
        }
        // acc += P * V  (4 hs-column blocks)
#pragma unroll
        for (int c = 0; c < 4; ++c) {
            AFrag bfrg;
            const bf16_t* p = Vt + (c * 16 + l16) * VLDS_STRIDE;
            bfrg.u[0] = *(const uint4*)(p + 8 * half);
            bfrg.u[1] = *(const uint4*)(p + 16 + 8 * half);
            acc[c] = __builtin_amdgcn_wmma_f32_16x16x32_bf16(
                false, pf.v, false, bfrg.v, (short)0, acc[c], false, false);
        }
    }

    float rl[8];
#pragma unroll
    for (int r = 0; r < 8; ++r) rl[r] = 1.0f / l_i[r];

    bf16_t* obase = Out + ((size_t)b * 2048 + q0) * 768 + h * 64;
#pragma unroll
    for (int c = 0; c < 4; ++c)
#pragma unroll
        for (int r = 0; r < 8; ++r)
            obase[(size_t)(r + 8 * half) * 768 + c * 16 + l16] = (bf16_t)(acc[c][r] * rl[r]);
}

// ---------------------------------------------------------------------------
// Launcher
// ---------------------------------------------------------------------------
extern "C" void kernel_launch(void* const* d_in, const int* in_sizes, int n_in,
                              void* d_out, int out_size, void* d_ws, size_t ws_size,
                              hipStream_t stream)
{
    (void)in_sizes; (void)n_in; (void)out_size; (void)ws_size;

    const float* x      = (const float*)d_in[0];
    const float* wq     = (const float*)d_in[1];
    const float* wk     = (const float*)d_in[2];
    const float* wv     = (const float*)d_in[3];
    const float* wo     = (const float*)d_in[4];
    const float* w1     = (const float*)d_in[5];
    const float* b1     = (const float*)d_in[6];
    const float* w2     = (const float*)d_in[7];
    const float* b2     = (const float*)d_in[8];
    const float* gamma1 = (const float*)d_in[9];
    const float* beta1  = (const float*)d_in[10];
    const float* gamma2 = (const float*)d_in[11];
    const float* beta2  = (const float*)d_in[12];
    float* out = (float*)d_out;

    const int BS = 4 * 2048;   // 8192 rows
    const int D = 768, HMLP = 3072;

    char* ws = (char*)d_ws;
    size_t off = 0;
    auto take = [&](size_t bytes) -> char* {
        char* p = ws + off;
        off += (bytes + 255) & ~(size_t)255;
        return p;
    };
    // all weights stored TRANSPOSED bf16: [N][K]
    bf16_t* wq_b  = (bf16_t*)take((size_t)D * D * 2);
    bf16_t* wk_b  = (bf16_t*)take((size_t)D * D * 2);
    bf16_t* wv_b  = (bf16_t*)take((size_t)D * D * 2);
    bf16_t* wo_b  = (bf16_t*)take((size_t)D * D * 2);
    bf16_t* w1_b  = (bf16_t*)take((size_t)D * HMLP * 2);   // [3072][768]
    bf16_t* w2_b  = (bf16_t*)take((size_t)HMLP * D * 2);   // [768][3072]
    bf16_t* xn    = (bf16_t*)take((size_t)BS * D * 2);
    bf16_t* qb    = (bf16_t*)take((size_t)BS * D * 2);
    bf16_t* kb    = (bf16_t*)take((size_t)BS * D * 2);
    bf16_t* vb    = (bf16_t*)take((size_t)BS * D * 2);
    bf16_t* attnb = (bf16_t*)take((size_t)BS * D * 2);
    float*  x1    = (float*) take((size_t)BS * D * 4);
    bf16_t* hb    = qb;  // reuse q/k/v/attn region (50.3MB) for h (50.3MB)

    // weight transpose-casts
    {
        int n = D * D;
        cast_transpose_f32_bf16<<<(n + 255) / 256, 256, 0, stream>>>(wq, wq_b, D, D);
        cast_transpose_f32_bf16<<<(n + 255) / 256, 256, 0, stream>>>(wk, wk_b, D, D);
        cast_transpose_f32_bf16<<<(n + 255) / 256, 256, 0, stream>>>(wv, wv_b, D, D);
        cast_transpose_f32_bf16<<<(n + 255) / 256, 256, 0, stream>>>(wo, wo_b, D, D);
        int n2 = D * HMLP;
        cast_transpose_f32_bf16<<<(n2 + 255) / 256, 256, 0, stream>>>(w1, w1_b, D, HMLP);
        cast_transpose_f32_bf16<<<(n2 + 255) / 256, 256, 0, stream>>>(w2, w2_b, HMLP, D);
    }

    const dim3 g768(BS / BM, D / BN);       // 64 x 6
    const dim3 g3072(BS / BM, HMLP / BN);   // 64 x 24

    // pre-LN attention (Q pre-scaled by 1/sqrt(hs) in its GEMM epilogue)
    layernorm_kernel<<<BS, 256, 0, stream>>>(x, gamma1, beta1, xn);
    gemm_bf16_kernel<false, false, false, true><<<g768, 256, 0, stream>>>(
        xn, wq_b, nullptr, nullptr, qb, BS, D, D, 0.125f);
    gemm_bf16_kernel<false, false, false, true><<<g768, 256, 0, stream>>>(
        xn, wk_b, nullptr, nullptr, kb, BS, D, D, 1.0f);
    gemm_bf16_kernel<false, false, false, true><<<g768, 256, 0, stream>>>(
        xn, wv_b, nullptr, nullptr, vb, BS, D, D, 1.0f);
    attention_kernel<<<(4 * 12 * 128) / ATT_WPB, 32 * ATT_WPB, 0, stream>>>(
        qb, kb, vb, attnb);
    gemm_bf16_kernel<false, false, true, false><<<g768, 256, 0, stream>>>(
        attnb, wo_b, nullptr, x, x1, BS, D, D, 1.0f);

    // pre-LN MLP
    layernorm_kernel<<<BS, 256, 0, stream>>>(x1, gamma2, beta2, xn);
    gemm_bf16_kernel<true, true, false, true><<<g3072, 256, 0, stream>>>(
        xn, w1_b, b1, nullptr, hb, BS, HMLP, D, 1.0f);
    gemm_bf16_kernel<true, false, true, false><<<g768, 256, 0, stream>>>(
        hb, w2_b, b2, x1, out, BS, D, HMLP, 1.0f);
}
